// ActorCriticTransformer_80796924772860
// MI455X (gfx1250) — compile-verified, matrix operand order checked by reading, not run on hardware
//
#include <hip/hip_runtime.h>
#include <hip/hip_bf16.h>

// ---------------------------------------------------------------------------
// GATv2 actor-critic for MI455X (gfx1250, wave32).
// GEMMs use v_wmma_f32_16x16x32_bf16 (bf16 operands, f32 accumulate).
// Edge softmax uses 3-pass atomics (max / exp-sum / normalize).
// ---------------------------------------------------------------------------

typedef __bf16 bf16;
typedef __attribute__((ext_vector_type(16))) bf16  v16bf;
typedef __attribute__((ext_vector_type(8)))  bf16  v8bf;
typedef __attribute__((ext_vector_type(8)))  float v8f;

#define NN   20000
#define DD   128
#define CC   128
#define HH   4
#define HCC  512
#define EE   320000
#define ETOT (EE + NN)
#define NEG_SLOPE 0.2f

// ----------------------------- small helpers ------------------------------

__device__ __forceinline__ float waveSum(float v) {
#pragma unroll
  for (int o = 16; o > 0; o >>= 1) v += __shfl_xor(v, o, 32);
  return v;
}

// float atomic max via signed/unsigned integer ordering trick
__device__ __forceinline__ void atomicMaxF(float* addr, float val) {
  if (val >= 0.0f) atomicMax((int*)addr, __float_as_int(val));
  else             atomicMin((unsigned int*)addr, (unsigned int)__float_as_int(val));
}

__global__ void fill_f32(float* __restrict__ p, float v, int n) {
  int i = blockIdx.x * blockDim.x + threadIdx.x;
  if (i < n) p[i] = v;
}

__global__ void cvt_bf16(const float* __restrict__ s, bf16* __restrict__ d, int n) {
  int i = blockIdx.x * blockDim.x + threadIdx.x;
  if (i < n) d[i] = (bf16)s[i];
}

// w[128][ncols] (row-major fp32) -> wt[ncols][128] (K-major bf16)
__global__ void transpose_cvt(const float* __restrict__ w, bf16* __restrict__ wt, int ncols) {
  int i = blockIdx.x * blockDim.x + threadIdx.x;
  int total = DD * ncols;
  if (i >= total) return;
  int k = i / ncols, n = i % ncols;
  wt[n * DD + k] = (bf16)w[i];
}

// ------------------------------- WMMA GEMM --------------------------------
// out[NN x NCOLS] = act(A[NN x 128] (bf16) @ W[128 x NCOLS] + bias)
// Wt is W transposed to [NCOLS][128] bf16 so both fragments load contiguously.
// One wave computes one 16x16 tile; K-loop = 4 x v_wmma_f32_16x16x32_bf16.

template <int NCOLS, bool RELU, bool WB16>
__global__ __launch_bounds__(256) void gemm_wmma(
    const bf16* __restrict__ A, const bf16* __restrict__ Wt,
    const float* __restrict__ bias, float* __restrict__ out,
    bf16* __restrict__ outb) {
  int gid  = blockIdx.x * blockDim.x + threadIdx.x;
  int wave = gid >> 5;
  int lane = gid & 31;
  const int tiles_n = NCOLS / 16;
  const int tiles_total = (NN / 16) * tiles_n;
  if (wave >= tiles_total) return;
  int tm = wave / tiles_n;
  int tn = wave % tiles_n;

  int arow = tm * 16 + (lane & 15);
  int bcol = tn * 16 + (lane & 15);
  int koff = (lane < 16) ? 0 : 8;

  const bf16* ap = A  + (size_t)arow * DD + koff;
  const bf16* bp = Wt + (size_t)bcol * DD + koff;

  v8f c = {};
#pragma unroll
  for (int k0 = 0; k0 < DD; k0 += 32) {
    v8bf alo = *(const v8bf*)(ap + k0);
    v8bf ahi = *(const v8bf*)(ap + k0 + 16);
    v8bf blo = *(const v8bf*)(bp + k0);
    v8bf bhi = *(const v8bf*)(bp + k0 + 16);
    v16bf a, b;
#pragma unroll
    for (int i = 0; i < 8; ++i) {
      a[i] = alo[i]; a[i + 8] = ahi[i];
      b[i] = blo[i]; b[i + 8] = bhi[i];
    }
    c = __builtin_amdgcn_wmma_f32_16x16x32_bf16(false, a, false, b,
                                                (short)0, c, false, false);
  }

  float bv = bias[bcol];
  int rbase = tm * 16 + ((lane < 16) ? 0 : 8);
#pragma unroll
  for (int r = 0; r < 8; ++r) {
    float v = c[r] + bv;
    if (RELU) v = fmaxf(v, 0.0f);
    size_t o = (size_t)(rbase + r) * NCOLS + bcol;
    out[o] = v;
    if (WB16) outb[o] = (bf16)v;
  }
}

// ------------------------------ edge kernels ------------------------------
// Pass A: score[e,h] = att_h . leaky_relu(xl[src]+xr[dst]); atomic max per (dst,h)
__global__ __launch_bounds__(256) void edge_score(
    const float* __restrict__ xl, const float* __restrict__ xr,
    const int* __restrict__ ei, const float* __restrict__ att,
    float* __restrict__ score, float* __restrict__ mmax) {
  int gid = blockIdx.x * blockDim.x + threadIdx.x;
  int e = gid >> 5, lane = gid & 31;
  if (e >= ETOT) return;
  int s, d;
  if (e < EE) { s = ei[e]; d = ei[EE + e]; } else { s = e - EE; d = s; }
  const float* pl = xl + (size_t)s * HCC;
  const float* pr = xr + (size_t)d * HCC;
#pragma unroll
  for (int h = 0; h < HH; ++h) {
    float4 a = *(const float4*)(pl + h * CC + lane * 4);
    float4 b = *(const float4*)(pr + h * CC + lane * 4);
    float4 w = *(const float4*)(att + h * CC + lane * 4);
    float v, t;
    v = a.x + b.x; v = (v > 0.0f) ? v : NEG_SLOPE * v; t  = v * w.x;
    v = a.y + b.y; v = (v > 0.0f) ? v : NEG_SLOPE * v; t += v * w.y;
    v = a.z + b.z; v = (v > 0.0f) ? v : NEG_SLOPE * v; t += v * w.z;
    v = a.w + b.w; v = (v > 0.0f) ? v : NEG_SLOPE * v; t += v * w.w;
    t = waveSum(t);
    if (lane == 0) {
      score[(size_t)e * HH + h] = t;
      atomicMaxF(&mmax[d * HH + h], t);
    }
  }
}

// Pass B: w = exp(score-m); denom[dst,h]+=w; acc[dst,h,:] += w*xl[src,h,:]
__global__ __launch_bounds__(256) void edge_accum(
    const float* __restrict__ xl, const float* __restrict__ score,
    const float* __restrict__ mmax, const int* __restrict__ ei,
    float* __restrict__ acc, float* __restrict__ denom) {
  int gid = blockIdx.x * blockDim.x + threadIdx.x;
  int e = gid >> 5, lane = gid & 31;
  if (e >= ETOT) return;
  int s, d;
  if (e < EE) { s = ei[e]; d = ei[EE + e]; } else { s = e - EE; d = s; }
#pragma unroll
  for (int h = 0; h < HH; ++h) {
    float wgt = __expf(score[(size_t)e * HH + h] - mmax[d * HH + h]);
    if (lane == 0) atomicAdd(&denom[d * HH + h], wgt);
    float4 a = *(const float4*)(xl + (size_t)s * HCC + h * CC + lane * 4);
    float* pa = acc + (size_t)d * HCC + h * CC + lane * 4;
    atomicAdd(pa + 0, wgt * a.x);
    atomicAdd(pa + 1, wgt * a.y);
    atomicAdd(pa + 2, wgt * a.z);
    atomicAdd(pa + 3, wgt * a.w);
  }
}

// Pass C: out[n,c] = mean_h(acc[n,h,c]/(denom[n,h]+eps)) + bias[c]
__global__ void gat_finalize(const float* __restrict__ acc,
                             const float* __restrict__ denom,
                             const float* __restrict__ bias,
                             float* __restrict__ out) {
  int i = blockIdx.x * blockDim.x + threadIdx.x;
  if (i >= NN * CC) return;
  int n = i / CC, c = i % CC;
  float s = 0.0f;
#pragma unroll
  for (int h = 0; h < HH; ++h)
    s += acc[(size_t)n * HCC + h * CC + c] / (denom[n * HH + h] + 1e-16f);
  out[i] = s * (1.0f / HH) + bias[c];
}

// out = relu(layernorm(in + res) * g + b); optional bf16 copy for next GEMM
__global__ __launch_bounds__(256) void res_ln_relu(
    const float* __restrict__ in, const float* __restrict__ res,
    const float* __restrict__ g, const float* __restrict__ b,
    float* __restrict__ out, bf16* __restrict__ outb) {
  int gid = blockIdx.x * blockDim.x + threadIdx.x;
  int n = gid >> 5, lane = gid & 31;
  if (n >= NN) return;
  float4 xi = *(const float4*)(in  + (size_t)n * CC + lane * 4);
  float4 xr = *(const float4*)(res + (size_t)n * CC + lane * 4);
  float4 x = make_float4(xi.x + xr.x, xi.y + xr.y, xi.z + xr.z, xi.w + xr.w);
  float s = waveSum(x.x + x.y + x.z + x.w);
  float mu = s * (1.0f / CC);
  float dx = x.x - mu, dy = x.y - mu, dz = x.z - mu, dw = x.w - mu;
  float sq = waveSum(dx * dx + dy * dy + dz * dz + dw * dw);
  float rstd = rsqrtf(sq * (1.0f / CC) + 1e-5f);
  float4 gg = *(const float4*)(g + lane * 4);
  float4 bb = *(const float4*)(b + lane * 4);
  float4 y;
  y.x = fmaxf(dx * rstd * gg.x + bb.x, 0.0f);
  y.y = fmaxf(dy * rstd * gg.y + bb.y, 0.0f);
  y.z = fmaxf(dz * rstd * gg.z + bb.z, 0.0f);
  y.w = fmaxf(dw * rstd * gg.w + bb.w, 0.0f);
  *(float4*)(out + (size_t)n * CC + lane * 4) = y;
  if (outb) {
    bf16* ob = outb + (size_t)n * CC + lane * 4;
    ob[0] = (bf16)y.x; ob[1] = (bf16)y.y; ob[2] = (bf16)y.z; ob[3] = (bf16)y.w;
  }
}

// out[n] = dot(x[n,:], w) + bias
__global__ __launch_bounds__(256) void head_dot(
    const float* __restrict__ x, const float* __restrict__ w,
    const float* __restrict__ bsc, float* __restrict__ out) {
  int gid = blockIdx.x * blockDim.x + threadIdx.x;
  int n = gid >> 5, lane = gid & 31;
  if (n >= NN) return;
  float4 a  = *(const float4*)(x + (size_t)n * CC + lane * 4);
  float4 ww = *(const float4*)(w + lane * 4);
  float t = waveSum(a.x * ww.x + a.y * ww.y + a.z * ww.z + a.w * ww.w);
  if (lane == 0) out[n] = t + bsc[0];
}

// -------------------------------- launcher --------------------------------

static inline unsigned nblk(long long threads) {
  return (unsigned)((threads + 255) / 256);
}

extern "C" void kernel_launch(void* const* d_in, const int* in_sizes, int n_in,
                              void* d_out, int out_size, void* d_ws, size_t ws_size,
                              hipStream_t stream) {
  const float* x     = (const float*)d_in[0];
  const int*   ei    = (const int*)d_in[1];
  const float* W_in  = (const float*)d_in[2];
  const float* b_in  = (const float*)d_in[3];
  const float* c1_Wl = (const float*)d_in[4];
  const float* c1_bl = (const float*)d_in[5];
  const float* c1_Wr = (const float*)d_in[6];
  const float* c1_br = (const float*)d_in[7];
  const float* c1_at = (const float*)d_in[8];
  const float* c1_b  = (const float*)d_in[9];
  const float* ln1_g = (const float*)d_in[10];
  const float* ln1_b = (const float*)d_in[11];
  const float* a_Wl  = (const float*)d_in[12];
  const float* a_bl  = (const float*)d_in[13];
  const float* a_Wr  = (const float*)d_in[14];
  const float* a_br  = (const float*)d_in[15];
  const float* a_at  = (const float*)d_in[16];
  const float* a_b   = (const float*)d_in[17];
  const float* aln_g = (const float*)d_in[18];
  const float* aln_b = (const float*)d_in[19];
  const float* ah_W  = (const float*)d_in[20];
  const float* ah_b  = (const float*)d_in[21];
  const float* k_Wl  = (const float*)d_in[22];
  const float* k_bl  = (const float*)d_in[23];
  const float* k_Wr  = (const float*)d_in[24];
  const float* k_br  = (const float*)d_in[25];
  const float* k_at  = (const float*)d_in[26];
  const float* k_b   = (const float*)d_in[27];
  const float* kln_g = (const float*)d_in[28];
  const float* kln_b = (const float*)d_in[29];
  const float* kh_W  = (const float*)d_in[30];
  const float* kh_b  = (const float*)d_in[31];
  float* dout = (float*)d_out;

  // ---- workspace carve-up (all sizes are multiples of 256 bytes) ----
  char* ws = (char*)d_ws;
  size_t off = 0;
  auto take = [&](size_t bytes) { char* p = ws + off; off += (bytes + 255) & ~(size_t)255; return p; };
  bf16*  xb    = (bf16*)take((size_t)NN * DD * 2);       // bf16(x)
  bf16*  hb    = (bf16*)take((size_t)NN * CC * 2);       // bf16 of current node feats
  float* h0    = (float*)take((size_t)NN * CC * 4);      // relu(x@W_in)
  float* h1    = (float*)take((size_t)NN * CC * 4);      // after layer-1 LN
  float* xl    = (float*)take((size_t)NN * HCC * 4);     // source transform
  float* xr    = (float*)take((size_t)NN * HCC * 4);     // dest transform / reused as acc
  float* score = (float*)take((size_t)ETOT * HH * 4);
  float* mmax  = (float*)take((size_t)NN * HH * 4);
  float* denom = (float*)take((size_t)NN * HH * 4);
  float* gat   = (float*)take((size_t)NN * CC * 4);      // GAT output / xa / xc
  bf16*  WinT  = (bf16*)take((size_t)DD * CC * 2);
  bf16*  WT_c1l = (bf16*)take((size_t)DD * HCC * 2);
  bf16*  WT_c1r = (bf16*)take((size_t)DD * HCC * 2);
  bf16*  WT_al  = (bf16*)take((size_t)DD * HCC * 2);
  bf16*  WT_ar  = (bf16*)take((size_t)DD * HCC * 2);
  bf16*  WT_kl  = (bf16*)take((size_t)DD * HCC * 2);
  bf16*  WT_kr  = (bf16*)take((size_t)DD * HCC * 2);
  (void)ws_size; (void)n_in; (void)in_sizes; (void)out_size;

  // ---- weight conversion (fp32 -> transposed bf16), input conversion ----
  transpose_cvt<<<nblk(DD * CC), 256, 0, stream>>>(W_in, WinT, CC);
  transpose_cvt<<<nblk(DD * HCC), 256, 0, stream>>>(c1_Wl, WT_c1l, HCC);
  transpose_cvt<<<nblk(DD * HCC), 256, 0, stream>>>(c1_Wr, WT_c1r, HCC);
  transpose_cvt<<<nblk(DD * HCC), 256, 0, stream>>>(a_Wl, WT_al, HCC);
  transpose_cvt<<<nblk(DD * HCC), 256, 0, stream>>>(a_Wr, WT_ar, HCC);
  transpose_cvt<<<nblk(DD * HCC), 256, 0, stream>>>(k_Wl, WT_kl, HCC);
  transpose_cvt<<<nblk(DD * HCC), 256, 0, stream>>>(k_Wr, WT_kr, HCC);
  cvt_bf16<<<nblk((long long)NN * DD), 256, 0, stream>>>(x, xb, NN * DD);

  // ---- h0 = relu(x @ W_in + b_in) (WMMA), also bf16 copy into hb ----
  {
    long long waves = (long long)(NN / 16) * (CC / 16);
    gemm_wmma<CC, true, true><<<nblk(waves * 32), 256, 0, stream>>>(
        xb, WinT, b_in, h0, hb);
  }

  // ---- one GATv2 layer: hin_b (bf16) -> gatout (fp32 N x C) ----
  auto run_gat = [&](const bf16* hin_b, const bf16* WlT, const float* bl,
                     const bf16* WrT, const float* br, const float* att,
                     const float* bconv, float* gatout) {
    long long waves = (long long)(NN / 16) * (HCC / 16);
    gemm_wmma<HCC, false, false><<<nblk(waves * 32), 256, 0, stream>>>(
        hin_b, WlT, bl, xl, (bf16*)nullptr);
    gemm_wmma<HCC, false, false><<<nblk(waves * 32), 256, 0, stream>>>(
        hin_b, WrT, br, xr, (bf16*)nullptr);
    fill_f32<<<nblk(NN * HH), 256, 0, stream>>>(mmax, -3.0e38f, NN * HH);
    edge_score<<<nblk((long long)ETOT * 32), 256, 0, stream>>>(
        xl, xr, ei, att, score, mmax);
    fill_f32<<<nblk((long long)NN * HCC), 256, 0, stream>>>(xr, 0.0f, NN * HCC);
    fill_f32<<<nblk(NN * HH), 256, 0, stream>>>(denom, 0.0f, NN * HH);
    edge_accum<<<nblk((long long)ETOT * 32), 256, 0, stream>>>(
        xl, score, mmax, ei, xr, denom);
    gat_finalize<<<nblk((long long)NN * CC), 256, 0, stream>>>(
        xr, denom, bconv, gatout);
  };

  // ---- layer 1 + residual LN, producing h1 (fp32) and hb (bf16) ----
  run_gat(hb, WT_c1l, c1_bl, WT_c1r, c1_br, c1_at, c1_b, gat);
  res_ln_relu<<<nblk((long long)NN * 32), 256, 0, stream>>>(
      gat, h0, ln1_g, ln1_b, h1, hb);

  // ---- actor branch ----
  run_gat(hb, WT_al, a_bl, WT_ar, a_br, a_at, a_b, gat);
  res_ln_relu<<<nblk((long long)NN * 32), 256, 0, stream>>>(
      gat, h1, aln_g, aln_b, gat, (bf16*)nullptr);
  head_dot<<<nblk((long long)NN * 32), 256, 0, stream>>>(gat, ah_W, ah_b, dout);

  // ---- critic branch ----
  run_gat(hb, WT_kl, k_bl, WT_kr, k_br, k_at, k_b, gat);
  res_ln_relu<<<nblk((long long)NN * 32), 256, 0, stream>>>(
      gat, h1, kln_g, kln_b, gat, (bf16*)nullptr);
  head_dot<<<nblk((long long)NN * 32), 256, 0, stream>>>(gat, kh_W, kh_b, dout + NN);
}
